// DeepGCN_84335977824420
// MI455X (gfx1250) — matrix operand order repaired and checked by reading, not compiled, not run on hardware
//
#include <hip/hip_runtime.h>
#include <hip/hip_bf16.h>

#define FF 256
#define HH 128
#define CC 64

typedef __attribute__((ext_vector_type(16))) __bf16 v16bf;
typedef __attribute__((ext_vector_type(8)))  float  v8f;

// ---------------------------------------------------------------------------
// Zero-fill (float4 granularity; all buffer sizes are multiples of 4 floats)
// ---------------------------------------------------------------------------
__global__ __launch_bounds__(256)
void zero4_kernel(float4* __restrict__ p, size_t n4) {
    size_t i = (size_t)blockIdx.x * blockDim.x + threadIdx.x;
    if (i < n4) p[i] = make_float4(0.f, 0.f, 0.f, 0.f);
}

// ---------------------------------------------------------------------------
// GEMM  S[N,HH] = X[N,FF] @ W[FF,HH]   (bf16x3 split-precision WMMA)
// grid.x = N/16 row tiles; 256 threads = 8 waves, wave w -> col tile w (HH=128)
// ---------------------------------------------------------------------------
__global__ __launch_bounds__(256)
void gemm_x_w_kernel(const float* __restrict__ X, const float* __restrict__ W,
                     float* __restrict__ S) {
    const int lane    = threadIdx.x & 31;
    const int wave    = threadIdx.x >> 5;
    const int rowBase = blockIdx.x * 16;
    const int colBase = wave * 16;
    const int half    = lane >> 4;     // 0: lanes 0-15, 1: lanes 16-31
    const int l15     = lane & 15;
    const int arow    = rowBase + l15; // A-matrix row held by this lane
    const int bcol    = colBase + l15; // B-matrix col held by this lane

    v8f acc = {};
    for (int k = 0; k < FF; k += 32) {
        // ---- A tile 16x32: lane half selects K-base (0 or 16); elem i -> K = kb+i
        const float* ap = X + (size_t)arow * FF + k + half * 16;
        float av[16];
        *(float4*)(&av[0])  = *(const float4*)(ap);
        *(float4*)(&av[4])  = *(const float4*)(ap + 4);
        *(float4*)(&av[8])  = *(const float4*)(ap + 8);
        *(float4*)(&av[12]) = *(const float4*)(ap + 12);
        v16bf ahi, alo;
#pragma unroll
        for (int i = 0; i < 16; ++i) {
            __bf16 h = (__bf16)av[i];
            ahi[i] = h;
            alo[i] = (__bf16)(av[i] - (float)h);
        }
        // ---- B tile 32x16: lane half selects K-base (0 or 16); elem i -> K = kb+i
        const float* bp = W + (size_t)(k + half * 16) * HH + bcol;
        v16bf bhi, blo;
#pragma unroll
        for (int i = 0; i < 16; ++i) {
            float f  = bp[(size_t)i * HH];
            __bf16 h = (__bf16)f;
            bhi[i] = h;
            blo[i] = (__bf16)(f - (float)h);
        }
        // ---- bf16x3: hi*hi + lo*hi + hi*lo  (lo*lo negligible)
        acc = __builtin_amdgcn_wmma_f32_16x16x32_bf16(false, ahi, false, bhi,
                                                      (short)0, acc, false, false);
        acc = __builtin_amdgcn_wmma_f32_16x16x32_bf16(false, alo, false, bhi,
                                                      (short)0, acc, false, false);
        acc = __builtin_amdgcn_wmma_f32_16x16x32_bf16(false, ahi, false, blo,
                                                      (short)0, acc, false, false);
    }
    // D layout: lane<16 -> M = r, lane>=16 -> M = 8+r; N = l15
#pragma unroll
    for (int r = 0; r < 8; ++r) {
        int m = r + half * 8;
        S[(size_t)(rowBase + m) * HH + bcol] = acc[r];
    }
}

// ---------------------------------------------------------------------------
// GEMM  O[N,CC] = relu(A[N,HH]) @ W[HH,CC]   (bf16x3 WMMA, relu on A load)
// grid.x = N/16 row tiles; 128 threads = 4 waves covering CC=64
// ---------------------------------------------------------------------------
__global__ __launch_bounds__(128)
void gemm_relu_w_kernel(const float* __restrict__ A, const float* __restrict__ W,
                        float* __restrict__ O) {
    const int lane    = threadIdx.x & 31;
    const int wave    = threadIdx.x >> 5;
    const int rowBase = blockIdx.x * 16;
    const int colBase = wave * 16;
    const int half    = lane >> 4;
    const int l15     = lane & 15;
    const int arow    = rowBase + l15;
    const int bcol    = colBase + l15;

    v8f acc = {};
    for (int k = 0; k < HH; k += 32) {
        const float* ap = A + (size_t)arow * HH + k + half * 16;
        float av[16];
        *(float4*)(&av[0])  = *(const float4*)(ap);
        *(float4*)(&av[4])  = *(const float4*)(ap + 4);
        *(float4*)(&av[8])  = *(const float4*)(ap + 8);
        *(float4*)(&av[12]) = *(const float4*)(ap + 12);
        v16bf ahi, alo;
#pragma unroll
        for (int i = 0; i < 16; ++i) {
            float f  = fmaxf(av[i], 0.f);        // fused ReLU
            __bf16 h = (__bf16)f;
            ahi[i] = h;
            alo[i] = (__bf16)(f - (float)h);
        }
        const float* bp = W + (size_t)(k + half * 16) * CC + bcol;
        v16bf bhi, blo;
#pragma unroll
        for (int i = 0; i < 16; ++i) {
            float f  = bp[(size_t)i * CC];
            __bf16 h = (__bf16)f;
            bhi[i] = h;
            blo[i] = (__bf16)(f - (float)h);
        }
        acc = __builtin_amdgcn_wmma_f32_16x16x32_bf16(false, ahi, false, bhi,
                                                      (short)0, acc, false, false);
        acc = __builtin_amdgcn_wmma_f32_16x16x32_bf16(false, alo, false, bhi,
                                                      (short)0, acc, false, false);
        acc = __builtin_amdgcn_wmma_f32_16x16x32_bf16(false, ahi, false, blo,
                                                      (short)0, acc, false, false);
    }
#pragma unroll
    for (int r = 0; r < 8; ++r) {
        int m = r + half * 8;
        O[(size_t)(rowBase + m) * CC + bcol] = acc[r];
    }
}

// ---------------------------------------------------------------------------
// SpMM scatter, D=128: one wave per edge; lane covers 4 contiguous floats.
// out[rows[e], :] += vals[e] * S[cols[e], :]   (L2-resident gather, atomic scatter)
// ---------------------------------------------------------------------------
__global__ __launch_bounds__(256)
void spmm128_kernel(const int* __restrict__ rows, const int* __restrict__ cols,
                    const float* __restrict__ vals, const float* __restrict__ S,
                    float* __restrict__ O, int E) {
    int e    = (int)(((size_t)blockIdx.x * blockDim.x + threadIdx.x) >> 5);
    int lane = threadIdx.x & 31;
    if (e >= E) return;
    int   r = rows[e];
    int   c = cols[e];
    float v = vals[e];
    float4 d = ((const float4*)(S + (size_t)c * 128))[lane];
    float* op = O + (size_t)r * 128 + lane * 4;
    atomicAdd(op + 0, v * d.x);
    atomicAdd(op + 1, v * d.y);
    atomicAdd(op + 2, v * d.z);
    atomicAdd(op + 3, v * d.w);
}

// SpMM scatter, D=64: one wave per edge; lane covers 2 contiguous floats.
__global__ __launch_bounds__(256)
void spmm64_kernel(const int* __restrict__ rows, const int* __restrict__ cols,
                   const float* __restrict__ vals, const float* __restrict__ S,
                   float* __restrict__ O, int E) {
    int e    = (int)(((size_t)blockIdx.x * blockDim.x + threadIdx.x) >> 5);
    int lane = threadIdx.x & 31;
    if (e >= E) return;
    int   r = rows[e];
    int   c = cols[e];
    float v = vals[e];
    float2 d = ((const float2*)(S + (size_t)c * 64))[lane];
    float* op = O + (size_t)r * 64 + lane * 2;
    atomicAdd(op + 0, v * d.x);
    atomicAdd(op + 1, v * d.y);
}

// ---------------------------------------------------------------------------
// Combine + bias + log_softmax over CC=64. One wave32 per node, 2 cols/lane.
// out = log_softmax( g*(h + b0) + (1-g)*(acc + b1 + b2) )
// ---------------------------------------------------------------------------
__global__ __launch_bounds__(256)
void combine_lsm_kernel(const float* __restrict__ Hm, const float* __restrict__ Acc,
                        const float* __restrict__ b0, const float* __restrict__ b1,
                        const float* __restrict__ b2, const float* __restrict__ gate,
                        float* __restrict__ out, int n) {
    int node = (int)(((size_t)blockIdx.x * blockDim.x + threadIdx.x) >> 5);
    int lane = threadIdx.x & 31;
    if (node >= n) return;
    float g  = gate[0];
    int   c0 = lane * 2, c1 = c0 + 1;
    const float* hm = Hm  + (size_t)node * CC;
    const float* ac = Acc + (size_t)node * CC;
    float v0 = g * (hm[c0] + b0[c0]) + (1.f - g) * (ac[c0] + b1[c0] + b2[c0]);
    float v1 = g * (hm[c1] + b0[c1]) + (1.f - g) * (ac[c1] + b1[c1] + b2[c1]);

    float m = fmaxf(v0, v1);
#pragma unroll
    for (int off = 16; off > 0; off >>= 1) m = fmaxf(m, __shfl_xor(m, off, 32));
    float s = expf(v0 - m) + expf(v1 - m);
#pragma unroll
    for (int off = 16; off > 0; off >>= 1) s += __shfl_xor(s, off, 32);
    float lse = m + logf(s);

    float* o = out + (size_t)node * CC;
    o[c0] = v0 - lse;
    o[c1] = v1 - lse;
}

// ---------------------------------------------------------------------------
extern "C" void kernel_launch(void* const* d_in, const int* in_sizes, int n_in,
                              void* d_out, int out_size, void* d_ws, size_t ws_size,
                              hipStream_t stream) {
    const float* x         = (const float*)d_in[0];
    const int*   mid_rows  = (const int*)d_in[1];
    const int*   mid_cols  = (const int*)d_in[2];
    const float* mid_vals  = (const float*)d_in[3];
    const int*   low_rows  = (const int*)d_in[4];
    const int*   low_cols  = (const int*)d_in[5];
    const float* low_vals  = (const float*)d_in[6];
    const int*   high_rows = (const int*)d_in[7];
    const int*   high_cols = (const int*)d_in[8];
    const float* high_vals = (const float*)d_in[9];
    const float* W_hid     = (const float*)d_in[10];
    const float* W_out     = (const float*)d_in[11];
    const float* b_out     = (const float*)d_in[12];
    const float* W_low     = (const float*)d_in[13];
    const float* W_low_out = (const float*)d_in[14];
    const float* b_low_out = (const float*)d_in[15];
    const float* W_high    = (const float*)d_in[16];
    const float* W_high_out= (const float*)d_in[17];
    const float* b_high_out= (const float*)d_in[18];
    const float* gate2     = (const float*)d_in[19];
    float*       out       = (float*)d_out;

    const int n     = in_sizes[0] / FF;   // 100000
    const int E_mid = in_sizes[1];
    const int E_low = in_sizes[4];
    const int E_high= in_sizes[7];

    // workspace: sup[N*HH] | agg[N*HH] | hid2[N*CC] | hmid[N*CC] | acc[N*CC]
    float* sup  = (float*)d_ws;
    float* agg  = sup  + (size_t)n * HH;
    float* hid2 = agg  + (size_t)n * HH;
    float* hmid = hid2 + (size_t)n * CC;
    float* acc  = hmid + (size_t)n * CC;

    const int rowTiles = n / 16;                     // 6250
    const size_t nH4 = (size_t)n * HH / 4;
    const size_t nC4 = (size_t)n * CC / 4;
    const int zgH = (int)((nH4 + 255) / 256);
    const int zgC = (int)((nC4 + 255) / 256);
    const int spM = (E_mid  + 7) / 8;                // 8 waves / 256-thread block
    const int spL = (E_low  + 7) / 8;
    const int spH = (E_high + 7) / 8;
    const int cwB = (n + 7) / 8;

    // shared accumulator for (x2 + xh)
    zero4_kernel<<<zgC, 256, 0, stream>>>((float4*)acc, nC4);

    // ---- MID path: h = spmm(mid, relu(spmm(mid, x@W_hid)) @ W_out)
    gemm_x_w_kernel<<<rowTiles, 256, 0, stream>>>(x, W_hid, sup);
    zero4_kernel<<<zgH, 256, 0, stream>>>((float4*)agg, nH4);
    spmm128_kernel<<<spM, 256, 0, stream>>>(mid_rows, mid_cols, mid_vals, sup, agg, E_mid);
    gemm_relu_w_kernel<<<rowTiles, 128, 0, stream>>>(agg, W_out, hid2);
    zero4_kernel<<<zgC, 256, 0, stream>>>((float4*)hmid, nC4);
    spmm64_kernel<<<spM, 256, 0, stream>>>(mid_rows, mid_cols, mid_vals, hid2, hmid, E_mid);

    // ---- LOW path -> acc
    gemm_x_w_kernel<<<rowTiles, 256, 0, stream>>>(x, W_low, sup);
    zero4_kernel<<<zgH, 256, 0, stream>>>((float4*)agg, nH4);
    spmm128_kernel<<<spL, 256, 0, stream>>>(low_rows, low_cols, low_vals, sup, agg, E_low);
    gemm_relu_w_kernel<<<rowTiles, 128, 0, stream>>>(agg, W_low_out, hid2);
    spmm64_kernel<<<spL, 256, 0, stream>>>(low_rows, low_cols, low_vals, hid2, acc, E_low);

    // ---- HIGH path -> acc (accumulates on top of LOW: reference sums x2+xh)
    gemm_x_w_kernel<<<rowTiles, 256, 0, stream>>>(x, W_high, sup);
    zero4_kernel<<<zgH, 256, 0, stream>>>((float4*)agg, nH4);
    spmm128_kernel<<<spH, 256, 0, stream>>>(high_rows, high_cols, high_vals, sup, agg, E_high);
    gemm_relu_w_kernel<<<rowTiles, 128, 0, stream>>>(agg, W_high_out, hid2);
    spmm64_kernel<<<spH, 256, 0, stream>>>(high_rows, high_cols, high_vals, hid2, acc, E_high);

    // ---- gate-combine + bias + log_softmax
    combine_lsm_kernel<<<cwB, 256, 0, stream>>>(hmid, acc, b_out, b_low_out,
                                                b_high_out, gate2, out, n);
}
